// WindowAttention_22969485099057
// MI455X (gfx1250) — compile-verified, matrix operand order checked by reading, not run on hardware
//
#include <hip/hip_runtime.h>

// ---------------------------------------------------------------------------
// MI455X (gfx1250) WMMA implementation of windowed attention + FFT mixer.
// All GEMMs use v_wmma_f32_16x16x32_f16 (f16 in, f32 accumulate).
// ---------------------------------------------------------------------------

typedef __attribute__((ext_vector_type(16))) _Float16 v16h;
typedef __attribute__((ext_vector_type(8)))  _Float16 v8h;
typedef __attribute__((ext_vector_type(8)))  float    v8f;

#define NTOK 49
#define CDIM 192
#define TDIM 576           // 3*CDIM
#define NHEAD 6
#define HDIM 32
#define NWIN 2048
#define ATT_SCALE 0.17677669529663688f   // 32^-0.5

// ---- workspace layout (bytes, 256-aligned) ----
#define WQKV_OFF 0                       // 576*192 f16  = 221184
#define W1_OFF   221184                  // 192*192 f16  =  73728
#define W2_OFF   294912
#define WP_OFF   368640
#define M_OFF    442368                  // 64*64 f16    =   8192
#define BIAS_OFF 450560                  // 6*49*49 f32  =  57624
#define PRE_OFF  508416                  // 2048*49*576 f16 = 115605504
#define Y_OFF    (PRE_OFF + (size_t)NWIN * NTOK * TDIM * 2)

// ---------------------------------------------------------------------------
// Fragment load: 16x32 operand tile from a row-major f16 matrix (ld elems/row).
// Lane l: row = l&15, kb = (l>=16)?8:0; elements K = kb..kb+7 and kb+16..kb+23.
// Lowered to two 16B loads per lane.
// ---------------------------------------------------------------------------
__device__ __forceinline__ v16h ldfrag(const _Float16* m, int ld, int lane) {
  const _Float16* p = m + (lane & 15) * ld + ((lane >> 4) << 3);
  v16h f;
#pragma unroll
  for (int i = 0; i < 8; ++i) f[i] = p[i];
#pragma unroll
  for (int i = 0; i < 8; ++i) f[8 + i] = p[16 + i];
  return f;
}

__device__ __forceinline__ v8f wmma16(v16h a, v16h b, v8f c) {
  return __builtin_amdgcn_wmma_f32_16x16x32_f16(false, a, false, b, (short)0, c,
                                                false, false);
}

// ---------------------------------------------------------------------------
// Init kernels
// ---------------------------------------------------------------------------
__global__ void initWeightsK(const float* __restrict__ qkv_w,
                             const float* __restrict__ q1w,
                             const float* __restrict__ q2w,
                             const float* __restrict__ pw,
                             _Float16* __restrict__ Wqkv, _Float16* __restrict__ W1,
                             _Float16* __restrict__ W2, _Float16* __restrict__ Wp) {
  int i = blockIdx.x * blockDim.x + threadIdx.x;
  if (i < TDIM * CDIM) Wqkv[i] = (_Float16)qkv_w[i];
  if (i < CDIM * CDIM) {
    W1[i] = (_Float16)q1w[i];
    W2[i] = (_Float16)q2w[i];
    Wp[i] = (_Float16)pw[i];
  }
}

// 64x64 zero-padded cosine matrix: M[n][m] = cos(2*pi*(u*i + v*j)/7),
// n = u*7+v (freq), m = i*7+j (space).  Symmetric.
__global__ void initMK(_Float16* __restrict__ Mg) {
  int i = blockIdx.x * blockDim.x + threadIdx.x;
  if (i >= 64 * 64) return;
  int n = i >> 6, m = i & 63;
  float val = 0.f;
  if (n < NTOK && m < NTOK) {
    int u = n / 7, v = n % 7, ii = m / 7, jj = m % 7;
    int ph = (u * ii + v * jj) % 7;
    val = cosf(6.283185307179586f * (float)ph / 7.0f);
  }
  Mg[i] = (_Float16)val;
}

// biasT[h][n][m] = rpb_table[RPI[n][m]][h]
__global__ void initBiasK(const float* __restrict__ rpb, float* __restrict__ biasT) {
  int i = blockIdx.x * blockDim.x + threadIdx.x;
  if (i >= NTOK * NTOK) return;
  int n = i / NTOK, m = i % NTOK;
  int r1 = n / 7, c1 = n % 7, r2 = m / 7, c2 = m % 7;
  int idx = (r1 - r2 + 6) * 13 + (c1 - c2 + 6);
#pragma unroll
  for (int h = 0; h < NHEAD; ++h)
    biasT[((size_t)h * NTOK + n) * NTOK + m] = rpb[idx * NHEAD + h];
}

// ---------------------------------------------------------------------------
// Phase A: per window b -> qkv_pre (f16) and FFT-branch Y (f16, flat c*49+m)
// ---------------------------------------------------------------------------
__global__ __launch_bounds__(256) void phaseA(
    const float* __restrict__ x, const float* __restrict__ qkv_b,
    const _Float16* __restrict__ Wqkv, const _Float16* __restrict__ W1,
    const _Float16* __restrict__ W2, const _Float16* __restrict__ Mg,
    _Float16* __restrict__ pre, _Float16* __restrict__ Yg) {
  __shared__ _Float16 bufA[64 * CDIM];  // Xh(64x192) -> F(64x192) -> H2^T(192x64)
  __shared__ _Float16 bufB[CDIM * 64];  // X^T(192x64) -> h1(64x192)
  __shared__ _Float16 Ms[64 * 64];

  const int b = blockIdx.x;
  const int t = threadIdx.x;
  const int wave = t >> 5, lane = t & 31;
  const int mt = wave & 3;              // fixed output-row panel per wave

  const float* xb = x + (size_t)b * NTOK * CDIM;
  for (int i = t; i < 64 * (CDIM / 4); i += 256) {
    int n = i / (CDIM / 4), c = (i % (CDIM / 4)) * 4;
    float4 xv = make_float4(0.f, 0.f, 0.f, 0.f);
    if (n < NTOK) xv = *(const float4*)(xb + n * CDIM + c);
    _Float16 h0 = (_Float16)xv.x, h1 = (_Float16)xv.y, h2 = (_Float16)xv.z,
             h3 = (_Float16)xv.w;
    bufA[n * CDIM + c + 0] = h0;
    bufA[n * CDIM + c + 1] = h1;
    bufA[n * CDIM + c + 2] = h2;
    bufA[n * CDIM + c + 3] = h3;
    bufB[(c + 0) * 64 + n] = h0;
    bufB[(c + 1) * 64 + n] = h1;
    bufB[(c + 2) * 64 + n] = h2;
    bufB[(c + 3) * 64 + n] = h3;
  }
  for (int i = t; i < 64 * 64; i += 256) Ms[i] = Mg[i];
  __syncthreads();

  // ---- qkv_pre = X @ Wqkv^T + qkv_b : 49x576 -> global f16 ----
  {
    v16h af[6];
#pragma unroll
    for (int kk = 0; kk < 6; ++kk)
      af[kk] = ldfrag(bufA + mt * 16 * CDIM + kk * 32, CDIM, lane);
    _Float16* dst = pre + (size_t)b * NTOK * TDIM;
    for (int nt = (wave >> 2); nt < 36; nt += 2) {
      const _Float16* wb = Wqkv + (size_t)(nt * 16) * CDIM;
      if (nt + 2 < 36)
        __builtin_prefetch(Wqkv + (size_t)((nt + 2) * 16 + (lane & 15)) * CDIM +
                               ((lane >> 4) << 3), 0, 1);
      v16h bf[6];
#pragma unroll
      for (int kk = 0; kk < 6; ++kk) bf[kk] = ldfrag(wb + kk * 32, CDIM, lane);
      v8f acc = {};
#pragma unroll
      for (int kk = 0; kk < 6; ++kk) acc = wmma16(af[kk], bf[kk], acc);
      int col = nt * 16 + (lane & 15);
      int rb = mt * 16 + ((lane >> 4) << 3);
      float bias = qkv_b[col];
      if (mt < 3) {  // wave-uniform: rows rb..rb+7 all < 48
#pragma unroll
        for (int j = 0; j < 8; ++j)
          dst[(rb + j) * TDIM + col] = (_Float16)(acc[j] + bias);
      } else if (lane < 16) {  // mt==3: only row 48 valid (rb==48, j==0)
        dst[48 * TDIM + col] = (_Float16)(acc[0] + bias);
      }
    }
  }
  __syncthreads();

  // cosine-matrix A fragments: reused by the F stage and the Y stage
  v16h afm[2];
#pragma unroll
  for (int kk = 0; kk < 2; ++kk)
    afm[kk] = ldfrag(Ms + mt * 16 * 64 + kk * 32, 64, lane);

  // ---- F = M @ X -> bufA (64x192) ----
  for (int nt = (wave >> 2); nt < 12; nt += 2) {
    v16h bf[2];
#pragma unroll
    for (int kk = 0; kk < 2; ++kk)
      bf[kk] = ldfrag(bufB + (nt * 16) * 64 + kk * 32, 64, lane);
    v8f acc = {};
#pragma unroll
    for (int kk = 0; kk < 2; ++kk) acc = wmma16(afm[kk], bf[kk], acc);
    int col = nt * 16 + (lane & 15);
    int rb = mt * 16 + ((lane >> 4) << 3);
#pragma unroll
    for (int j = 0; j < 8; ++j) bufA[(rb + j) * CDIM + col] = (_Float16)acc[j];
  }
  __syncthreads();

  // ---- h1 = gelu(F @ W1^T) -> bufB (64x192) ----
  {
    v16h af[6];
#pragma unroll
    for (int kk = 0; kk < 6; ++kk)
      af[kk] = ldfrag(bufA + mt * 16 * CDIM + kk * 32, CDIM, lane);
    for (int nt = (wave >> 2); nt < 12; nt += 2) {
      v16h bf[6];
#pragma unroll
      for (int kk = 0; kk < 6; ++kk)
        bf[kk] = ldfrag(W1 + (size_t)(nt * 16) * CDIM + kk * 32, CDIM, lane);
      v8f acc = {};
#pragma unroll
      for (int kk = 0; kk < 6; ++kk) acc = wmma16(af[kk], bf[kk], acc);
      int col = nt * 16 + (lane & 15);
      int rb = mt * 16 + ((lane >> 4) << 3);
#pragma unroll
      for (int j = 0; j < 8; ++j) {
        float xv = acc[j];
        bufB[(rb + j) * CDIM + col] =
            (_Float16)(0.5f * xv * (1.f + erff(xv * 0.7071067811865475f)));
      }
    }
  }
  __syncthreads();

  // ---- H2^T = (h1 @ W2^T)^T -> bufA (192x64) ----
  {
    v16h af[6];
#pragma unroll
    for (int kk = 0; kk < 6; ++kk)
      af[kk] = ldfrag(bufB + mt * 16 * CDIM + kk * 32, CDIM, lane);
    for (int nt = (wave >> 2); nt < 12; nt += 2) {
      v16h bf[6];
#pragma unroll
      for (int kk = 0; kk < 6; ++kk)
        bf[kk] = ldfrag(W2 + (size_t)(nt * 16) * CDIM + kk * 32, CDIM, lane);
      v8f acc = {};
#pragma unroll
      for (int kk = 0; kk < 6; ++kk) acc = wmma16(af[kk], bf[kk], acc);
      int col = nt * 16 + (lane & 15);
      int rb = mt * 16 + ((lane >> 4) << 3);
#pragma unroll
      for (int j = 0; j < 8; ++j) bufA[col * 64 + rb + j] = (_Float16)acc[j];
    }
  }
  __syncthreads();

  // ---- Y = (1/49) * M @ H2 -> global f16, flat layout [c*49 + m] ----
  {
    _Float16* yb = Yg + (size_t)b * CDIM * NTOK;
    for (int nt = (wave >> 2); nt < 12; nt += 2) {
      v16h bf[2];
#pragma unroll
      for (int kk = 0; kk < 2; ++kk)
        bf[kk] = ldfrag(bufA + (nt * 16) * 64 + kk * 32, 64, lane);
      v8f acc = {};
#pragma unroll
      for (int kk = 0; kk < 2; ++kk) acc = wmma16(afm[kk], bf[kk], acc);
      int col = nt * 16 + (lane & 15);
      int rb = mt * 16 + ((lane >> 4) << 3);
      if (mt < 3) {
#pragma unroll
        for (int j = 0; j < 8; ++j)
          yb[col * NTOK + rb + j] = (_Float16)(acc[j] * (1.0f / 49.0f));
      } else if (lane < 16) {
        yb[col * NTOK + 48] = (_Float16)(acc[0] * (1.0f / 49.0f));
      }
    }
  }
}

// ---------------------------------------------------------------------------
// Phase B: depthwise conv (couples b-1,b,b+1) + Y add -> attention -> proj
// ---------------------------------------------------------------------------
__global__ __launch_bounds__(256) void phaseB(
    const _Float16* __restrict__ pre, const _Float16* __restrict__ Yg,
    const float* __restrict__ dw_w, const float* __restrict__ dw_b,
    const float* __restrict__ biasT, const _Float16* __restrict__ Wp,
    const float* __restrict__ proj_b, float* __restrict__ out) {
  extern __shared__ __align__(16) char smem[];
  _Float16* qh = (_Float16*)smem;        // [6][64][32]  (scaled q, zero-pad rows)
  _Float16* kh = qh + 6 * 64 * 32;       // [6][64][32]
  _Float16* vt = kh + 6 * 64 * 32;       // [6][32][64]  (v^T, zero-pad cols)
  _Float16* oh = vt + 6 * 32 * 64;       // [64][192]    merged heads
  _Float16* Pb = oh + 64 * CDIM;         // [64][64]     exp(S) (unnormalized)
  float* Sb = (float*)(Pb + 64 * 64);    // [64][64]     raw scores
  float* pmax = Sb + 64 * 64;            // [4][64]      partial row max
  float* psum = pmax + 4 * 64;           // [4][64]      partial row sums
  float* dwT = psum + 4 * 64;            // [9][576]     tap-major conv weights
  float* dwBl = dwT + 9 * TDIM;          // [576]

  const int b = blockIdx.x;
  const int t = threadIdx.x;
  const int wave = t >> 5, lane = t & 31;
  const int mt = wave & 3;

  for (int i = t; i < 3 * 6 * 64 * 32; i += 256) qh[i] = (_Float16)0.f;
  for (int i = t; i < TDIM * 9; i += 256) {
    int ch = i / 9, tap = i % 9;
    dwT[tap * TDIM + ch] = dw_w[i];
  }
  for (int i = t; i < TDIM; i += 256) dwBl[i] = dw_b[i];
  __syncthreads();

  // depthwise 3x3 conv over (window, token) + FFT-branch add -> q/k/v in LDS.
  // 8 channels per thread: 16B global loads, tap-major weights from LDS.
  {
    int cg = t % 72, n = t / 72;   // 72 = TDIM/8 channel groups
    while (n < NTOK) {
      int ch0 = cg * 8;
      float acc[8];
#pragma unroll
      for (int j = 0; j < 8; ++j) acc[j] = dwBl[ch0 + j];
#pragma unroll
      for (int db = -1; db <= 1; ++db) {
        int b2 = b + db;
        if (b2 < 0 || b2 >= NWIN) continue;
        const _Float16* pb2 = pre + (size_t)b2 * NTOK * TDIM + ch0;
#pragma unroll
        for (int dn = -1; dn <= 1; ++dn) {
          int n2 = n + dn;
          if (n2 < 0 || n2 >= NTOK) continue;
          v8h d8 = *(const v8h*)(pb2 + (size_t)n2 * TDIM);
          const float* wt = dwT + ((db + 1) * 3 + (dn + 1)) * TDIM + ch0;
#pragma unroll
          for (int j = 0; j < 8; ++j) acc[j] += wt[j] * (float)d8[j];
        }
      }
      int s = ch0 / CDIM, cc = ch0 % CDIM, h = cc >> 5, d0 = cc & 31;
      // reference reshape (B,C,49)->(B,6,49,32) is a flat reindex: h*1568+n*32+d
      v8h yv = *(const v8h*)(Yg + (size_t)b * CDIM * NTOK + h * (NTOK * HDIM) +
                             n * HDIM + d0);
      if (s == 0) {
        v8h r;
#pragma unroll
        for (int j = 0; j < 8; ++j)
          r[j] = (_Float16)((acc[j] + (float)yv[j]) * ATT_SCALE);
        *(v8h*)(qh + (h * 64 + n) * 32 + d0) = r;
      } else if (s == 1) {
        v8h r;
#pragma unroll
        for (int j = 0; j < 8; ++j) r[j] = (_Float16)(acc[j] + (float)yv[j]);
        *(v8h*)(kh + (h * 64 + n) * 32 + d0) = r;
      } else {
#pragma unroll
        for (int j = 0; j < 8; ++j)
          vt[(h * 32 + d0 + j) * 64 + n] = (_Float16)(acc[j] + (float)yv[j]);
      }
      n += 3; cg += 40;                 // advance by 256 work items
      if (cg >= 72) { cg -= 72; n += 1; }
    }
  }
  __syncthreads();

  for (int h = 0; h < NHEAD; ++h) {
    // S = q @ k^T : 4x4 token tiles, K = 32 (single WMMA per tile)
    {
      v16h aq = ldfrag(qh + (h * 64 + mt * 16) * 32, 32, lane);
#pragma unroll
      for (int nt = (wave >> 2); nt < 4; nt += 2) {
        v16h bk = ldfrag(kh + (h * 64 + nt * 16) * 32, 32, lane);
        v8f acc = {};
        acc = wmma16(aq, bk, acc);
        int col = nt * 16 + (lane & 15);
        int rb = mt * 16 + ((lane >> 4) << 3);
#pragma unroll
        for (int j = 0; j < 8; ++j) Sb[(rb + j) * 64 + col] = acc[j];
      }
    }
    __syncthreads();

    // parallel softmax: 64 rows x 4 column parts (13/13/13/10 cols each).
    {
      int r = t & 63, p = t >> 6;
      int m0 = p * 13, m1 = m0 + 13;
      if (m1 > NTOK) m1 = NTOK;
      float mx = -1e30f;
      if (r < NTOK) {
        const float* brow = biasT + ((size_t)h * NTOK + r) * NTOK;
        for (int m = m0; m < m1; ++m) mx = fmaxf(mx, Sb[r * 64 + m] + brow[m]);
      }
      pmax[p * 64 + r] = mx;
    }
    __syncthreads();
    {
      int r = t & 63, p = t >> 6;
      int m0 = p * 13, m1 = m0 + 13;
      if (m1 > NTOK) m1 = NTOK;
      float sum = 0.f;
      if (r < NTOK) {
        float mx = fmaxf(fmaxf(pmax[r], pmax[64 + r]),
                         fmaxf(pmax[128 + r], pmax[192 + r]));
        const float* brow = biasT + ((size_t)h * NTOK + r) * NTOK;
        for (int m = m0; m < m1; ++m) {
          float e = __expf(Sb[r * 64 + m] + brow[m] - mx);
          sum += e;
          Pb[r * 64 + m] = (_Float16)e;
        }
        if (p == 3)
          for (int m = NTOK; m < 64; ++m) Pb[r * 64 + m] = (_Float16)0.f;
      } else if (p == 0) {
        for (int m = 0; m < 64; ++m) Pb[r * 64 + m] = (_Float16)0.f;
      }
      psum[p * 64 + r] = sum;
    }
    __syncthreads();

    // O_h = (P @ v) / rowsum -> oh[:, h*32 .. h*32+31]  (8 tiles, 1 per wave)
    {
      int nt = wave >> 2;
      v16h af[2], bf[2];
#pragma unroll
      for (int kk = 0; kk < 2; ++kk)
        af[kk] = ldfrag(Pb + mt * 16 * 64 + kk * 32, 64, lane);
#pragma unroll
      for (int kk = 0; kk < 2; ++kk)
        bf[kk] = ldfrag(vt + (h * 32 + nt * 16) * 64 + kk * 32, 64, lane);
      v8f acc = {};
#pragma unroll
      for (int kk = 0; kk < 2; ++kk) acc = wmma16(af[kk], bf[kk], acc);
      int col = nt * 16 + (lane & 15);
      int rb = mt * 16 + ((lane >> 4) << 3);
#pragma unroll
      for (int j = 0; j < 8; ++j) {
        int r = rb + j;
        float rs = psum[r] + psum[64 + r] + psum[128 + r] + psum[192 + r];
        rs = (rs > 0.f) ? rs : 1.f;     // pad rows: 0/1 instead of 0/0
        oh[r * CDIM + h * HDIM + col] = (_Float16)(acc[j] / rs);
      }
    }
    __syncthreads();
  }

  // ---- out = oh @ Wp^T + proj_b -> global f32 ----
  {
    v16h af[6];
#pragma unroll
    for (int kk = 0; kk < 6; ++kk)
      af[kk] = ldfrag(oh + mt * 16 * CDIM + kk * 32, CDIM, lane);
    float* ob = out + (size_t)b * NTOK * CDIM;
    for (int nt = (wave >> 2); nt < 12; nt += 2) {
      v16h bf[6];
#pragma unroll
      for (int kk = 0; kk < 6; ++kk)
        bf[kk] = ldfrag(Wp + (size_t)(nt * 16) * CDIM + kk * 32, CDIM, lane);
      v8f acc = {};
#pragma unroll
      for (int kk = 0; kk < 6; ++kk) acc = wmma16(af[kk], bf[kk], acc);
      int col = nt * 16 + (lane & 15);
      int rb = mt * 16 + ((lane >> 4) << 3);
      float pbv = proj_b[col];
      if (mt < 3) {
#pragma unroll
        for (int j = 0; j < 8; ++j) ob[(rb + j) * CDIM + col] = acc[j] + pbv;
      } else if (lane < 16) {
        ob[48 * CDIM + col] = acc[0] + pbv;
      }
    }
  }
}

// ---------------------------------------------------------------------------
extern "C" void kernel_launch(void* const* d_in, const int* in_sizes, int n_in,
                              void* d_out, int out_size, void* d_ws, size_t ws_size,
                              hipStream_t stream) {
  (void)in_sizes; (void)n_in; (void)out_size; (void)ws_size;
  const float* x     = (const float*)d_in[0];
  const float* rpb   = (const float*)d_in[1];
  const float* qkv_w = (const float*)d_in[2];
  const float* qkv_b = (const float*)d_in[3];
  const float* dw_w  = (const float*)d_in[4];
  const float* dw_b  = (const float*)d_in[5];
  const float* q1w   = (const float*)d_in[6];
  const float* q2w   = (const float*)d_in[7];
  const float* pw    = (const float*)d_in[8];
  const float* pb    = (const float*)d_in[9];

  char* ws = (char*)d_ws;
  _Float16* Wqkv  = (_Float16*)(ws + WQKV_OFF);
  _Float16* W1    = (_Float16*)(ws + W1_OFF);
  _Float16* W2    = (_Float16*)(ws + W2_OFF);
  _Float16* Wp    = (_Float16*)(ws + WP_OFF);
  _Float16* Mg    = (_Float16*)(ws + M_OFF);
  float*    biasT = (float*)(ws + BIAS_OFF);
  _Float16* preB  = (_Float16*)(ws + PRE_OFF);
  _Float16* Yg    = (_Float16*)(ws + Y_OFF);

  initWeightsK<<<(TDIM * CDIM + 255) / 256, 256, 0, stream>>>(qkv_w, q1w, q2w, pw,
                                                              Wqkv, W1, W2, Wp);
  initMK<<<(64 * 64 + 255) / 256, 256, 0, stream>>>(Mg);
  initBiasK<<<(NTOK * NTOK + 255) / 256, 256, 0, stream>>>(rpb, biasT);

  phaseA<<<NWIN, 256, 0, stream>>>(x, qkv_b, Wqkv, W1, W2, Mg, preB, Yg);

  size_t smemB = (size_t)(6 * 64 * 32 * 2 + 6 * 64 * 32 + 6 * 32 * 64 + 64 * CDIM +
                          64 * 64) * sizeof(_Float16) +
                 (size_t)(64 * 64 + 4 * 64 + 4 * 64 + 9 * TDIM + TDIM) *
                     sizeof(float);
  phaseB<<<NWIN, 256, smemB, stream>>>(preB, Yg, dw_w, dw_b, biasT, Wp, pb,
                                       (float*)d_out);
}
